// MultiHeadAttention_31336081392015
// MI455X (gfx1250) — matrix operand order repaired and checked by reading, not compiled, not run on hardware
//
#include <hip/hip_runtime.h>
#include <hip/hip_bf16.h>

// MI455X / gfx1250 fused multi-head attention (compile-only target).
// f16 WMMA inputs, f32 accumulation; async LDS staging + double buffering
// in the attention kernel.

typedef __attribute__((ext_vector_type(16))) _Float16 v16h;
typedef __attribute__((ext_vector_type(8)))  float    v8f;
typedef __attribute__((ext_vector_type(4)))  unsigned u32x4;
typedef __attribute__((ext_vector_type(4)))  float    f32x4;

#define B_   32
#define N_   1024
#define D_   512
#define H_   8
#define DH_  64
#define TOK_ (B_ * N_)   // 32768

union V16H {
    v16h     v;
    u32x4    q[2];
    unsigned u[8];
    _Float16 h[16];
};

union H16 {
    _Float16 h[16];
    u32x4    q[2];
};

__device__ __forceinline__ v8f zero8() {
    v8f z = {0.f, 0.f, 0.f, 0.f, 0.f, 0.f, 0.f, 0.f};
    return z;
}

// CDNA5 async global->LDS copy, 16B per lane.  VDST = per-lane LDS byte
// offset, VADDR = per-lane 64-bit global address, GV mode (saddr = off).
// Tracked by ASYNCcnt.
__device__ __forceinline__ void async_ld16(unsigned lds_byte_off, const void* gsrc) {
    asm volatile("global_load_async_to_lds_b128 %0, %1, off"
                 :: "v"(lds_byte_off), "v"(gsrc)
                 : "memory");
}

template <int N>
__device__ __forceinline__ void wait_async() {
    asm volatile("s_wait_asynccnt %0" :: "n"(N) : "memory");
}

// Flat pointer -> LDS byte offset (flat->LDS aperture mapping truncates to
// addr[31:0]).
__device__ __forceinline__ unsigned lds_off(const void* p) {
    return (unsigned)(size_t)p;
}

// ---------------------------------------------------------------------------
// Kernel 1: QKV projection.  C[T, dout] = x[T, :] . W[dout, :] + b[dout]
// Tile: 128 tokens x 64 douts per 256-thread block; blockIdx.z selects Q/K/V.
// f16 outputs in head-permuted layouts (raw-reshape semantics of the ref):
//   q/k : [b][h][nn][dh]   h = n>>7, nn = (n&127)*8 + d/64, dh = d&63
//   v   : [b][h][dh][nn]   (transposed: P@V B-operand reads contiguous)
// Q gets the 1/sqrt(DH) score scale folded in.
// ---------------------------------------------------------------------------
__global__ __launch_bounds__(256) void qkv_proj_kernel(
    const float* __restrict__ x,
    const float* __restrict__ Wq, const float* __restrict__ bq,
    const float* __restrict__ Wk, const float* __restrict__ bk,
    const float* __restrict__ Wv, const float* __restrict__ bv,
    _Float16* __restrict__ qws, _Float16* __restrict__ kws,
    _Float16* __restrict__ vws)
{
    const int z = blockIdx.z;
    const float* __restrict__ W    = (z == 0) ? Wq : ((z == 1) ? Wk : Wv);
    const float* __restrict__ bias = (z == 0) ? bq : ((z == 1) ? bk : bv);

    __shared__ __align__(16) _Float16 xl[128 * 32]; // x tile [token][k]
    __shared__ __align__(16) _Float16 wl[64 * 32];  // W tile [dout][k] == B [n][k]

    const int t    = threadIdx.x;
    const int wave = t >> 5;
    const int lane = t & 31;
    const int lo   = lane & 15;
    const int g    = lane >> 4;

    const int Mbase = blockIdx.x * 128; // token base
    const int Nbase = blockIdx.y * 64;  // dout base

    v8f acc[4];
    #pragma unroll
    for (int i = 0; i < 4; ++i) acc[i] = zero8();

    for (int kc = 0; kc < 16; ++kc) {
        // ---- stage x tile (128x32 f32 -> f16), 16 elems/thread ----
        {
            const int r  = t >> 1;
            const int c0 = (t & 1) * 16;
            const float* src = x + (size_t)(Mbase + r) * D_ + kc * 32 + c0;
            H16 tmp;
            #pragma unroll
            for (int j = 0; j < 4; ++j) {
                f32x4 f = ((const f32x4*)src)[j];
                tmp.h[4 * j + 0] = (_Float16)f.x;
                tmp.h[4 * j + 1] = (_Float16)f.y;
                tmp.h[4 * j + 2] = (_Float16)f.z;
                tmp.h[4 * j + 3] = (_Float16)f.w;
            }
            ((u32x4*)&xl[r * 32 + c0])[0] = tmp.q[0];
            ((u32x4*)&xl[r * 32 + c0])[1] = tmp.q[1];
        }
        // ---- stage W tile (64x32 f32 -> f16), 8 elems/thread ----
        {
            const int r  = t >> 2;
            const int c0 = (t & 3) * 8;
            const float* src = W + (size_t)(Nbase + r) * D_ + kc * 32 + c0;
            H16 tmp;
            #pragma unroll
            for (int j = 0; j < 2; ++j) {
                f32x4 f = ((const f32x4*)src)[j];
                tmp.h[4 * j + 0] = (_Float16)f.x;
                tmp.h[4 * j + 1] = (_Float16)f.y;
                tmp.h[4 * j + 2] = (_Float16)f.z;
                tmp.h[4 * j + 3] = (_Float16)f.w;
            }
            ((u32x4*)&wl[r * 32 + c0])[0] = tmp.q[0];
        }
        __syncthreads();

        // ---- A operand: x rows, contiguous k order (k = 16*g + s) ----
        V16H a;
        const _Float16* arow = &xl[(wave * 16 + lo) * 32];
        #pragma unroll
        for (int v = 0; v < 8; ++v) {
            const int K0 = 2 * v + 16 * g;
            a.u[v] = *(const unsigned*)&arow[K0];
        }
        // ---- 4 B operands + WMMAs ----
        #pragma unroll
        for (int tn = 0; tn < 4; ++tn) {
            V16H b;
            const _Float16* brow = &wl[(tn * 16 + lo) * 32 + 16 * g];
            b.q[0] = ((const u32x4*)brow)[0];
            b.q[1] = ((const u32x4*)brow)[1];
            acc[tn] = __builtin_amdgcn_wmma_f32_16x16x32_f16(
                false, a.v, false, b.v, (short)0, acc[tn], false, false);
        }
        __syncthreads();
    }

    // ---- epilogue: bias, Q-scale, f16 convert, permuted scatter ----
    const float qscale = 0.125f; // 1/sqrt(64)
    #pragma unroll
    for (int tn = 0; tn < 4; ++tn) {
        const int   dout = Nbase + tn * 16 + lo;
        const float bval = bias[dout];
        #pragma unroll
        for (int i = 0; i < 8; ++i) {
            const int T = Mbase + wave * 16 + i + 8 * g; // global token
            float val = acc[tn][i] + bval;
            if (z == 0) val *= qscale;
            const _Float16 hv = (_Float16)val;
            const int b  = T >> 10;
            const int n  = T & 1023;
            const int h  = n >> 7;
            const int nn = ((n & 127) << 3) | (dout >> 6);
            const int dh = dout & 63;
            const int bh = b * H_ + h;
            if (z == 0)      qws[((size_t)bh * 1024 + nn) * 64 + dh] = hv;
            else if (z == 1) kws[((size_t)bh * 1024 + nn) * 64 + dh] = hv;
            else             vws[((size_t)bh * 64 + dh) * 1024 + nn] = hv;
        }
    }
}

// ---------------------------------------------------------------------------
// Kernel 2: flash attention per (b*h).  Q A-operands live in registers for
// the whole key loop.  32-key K / V^T blocks are DMA'd into double-buffered
// LDS with global_load_async_to_lds_b128 (ASYNCcnt), prefetching block kb+1
// under block kb's WMMAs.  Online softmax via 16-lane shuffle reductions;
// P re-laid out C->A through per-wave LDS.
// ---------------------------------------------------------------------------
__global__ __launch_bounds__(256) void attn_kernel(
    const _Float16* __restrict__ qws, const _Float16* __restrict__ kws,
    const _Float16* __restrict__ vws, float* __restrict__ out)
{
    __shared__ __align__(16) _Float16 klds[2][32 * 64];  // [buf][key][dh]
    __shared__ __align__(16) _Float16 vlds[2][64 * 32];  // [buf][dh][key]
    __shared__ __align__(16) _Float16 plds[8 * 16 * 32]; // per-wave P [m][key]

    const int t    = threadIdx.x;
    const int wave = t >> 5;
    const int lane = t & 31;
    const int lo   = lane & 15;
    const int g    = lane >> 4;

    const int bh    = blockIdx.x;                  // 0..255
    const int qbase = blockIdx.y * 128 + wave * 16;
    const int m     = qbase + lo;                  // this lane's A-row

    // Per-thread source/dest slots for the async staging (16B each).
    const _Float16* kbase = kws + (size_t)bh * 65536 + t * 8;  // + kb*2048
    const int vr = t >> 2, vp = t & 3;
    const _Float16* vbase = vws + (size_t)bh * 65536 + vr * 1024 + vp * 8; // + kb*32
    const unsigned kdst[2] = { lds_off(&klds[0][t * 8]), lds_off(&klds[1][t * 8]) };
    const unsigned vdst[2] = { lds_off(&vlds[0][vr * 32 + vp * 8]),
                               lds_off(&vlds[1][vr * 32 + vp * 8]) };

    // Q A-operands for dh chunks 0..31 and 32..63 stay in registers.
    V16H aq[2];
    const _Float16* qrow = qws + ((size_t)bh * 1024 + m) * 64;
    #pragma unroll
    for (int c = 0; c < 2; ++c) {
        #pragma unroll
        for (int v = 0; v < 8; ++v) {
            const int K0 = 32 * c + 2 * v + 16 * g;
            aq[c].u[v] = *(const unsigned*)&qrow[K0];
        }
    }

    v8f acc[4];
    #pragma unroll
    for (int d = 0; d < 4; ++d) acc[d] = zero8();
    float mrow[8], lrow[8];
    #pragma unroll
    for (int i = 0; i < 8; ++i) { mrow[i] = -3.0e38f; lrow[i] = 0.f; }

    _Float16* pw = &plds[wave * 16 * 32];

    // Prologue: kick off block 0 into buffer 0.
    async_ld16(kdst[0], kbase);
    async_ld16(vdst[0], vbase);

    for (int kb = 0; kb < 32; ++kb) {
        const int cur = kb & 1;

        // Prefetch block kb+1 into the other buffer.  Safe: the trailing
        // barrier of iteration kb-1 guarantees nobody still reads it.
        if (kb < 31) {
            async_ld16(kdst[cur ^ 1], kbase + (size_t)(kb + 1) * 2048);
            async_ld16(vdst[cur ^ 1], vbase + (kb + 1) * 32);
            wait_async<2>();   // current block done; 2 fresh ones in flight
        } else {
            wait_async<0>();
        }
        __syncthreads();       // all waves' copies of block kb complete

        const _Float16* kl = klds[cur];
        const _Float16* vl = vlds[cur];

        // ---- scores: 2 tiles of 16 keys, K-dim = dh = 64 (2 chunks) ----
        v8f S[2];
        S[0] = zero8(); S[1] = zero8();
        #pragma unroll
        for (int tt = 0; tt < 2; ++tt) {
            #pragma unroll
            for (int c = 0; c < 2; ++c) {
                V16H bk_;
                const _Float16* kr = &kl[(tt * 16 + lo) * 64 + 32 * c + 16 * g];
                bk_.q[0] = ((const u32x4*)kr)[0];
                bk_.q[1] = ((const u32x4*)kr)[1];
                S[tt] = __builtin_amdgcn_wmma_f32_16x16x32_f16(
                    false, aq[c].v, false, bk_.v, (short)0, S[tt], false, false);
            }
        }

        // ---- online softmax (rows live on 16-lane groups) ----
        #pragma unroll
        for (int i = 0; i < 8; ++i) {
            float cmb = fmaxf(S[0][i], S[1][i]);
            cmb = fmaxf(cmb, __shfl_xor(cmb, 1));
            cmb = fmaxf(cmb, __shfl_xor(cmb, 2));
            cmb = fmaxf(cmb, __shfl_xor(cmb, 4));
            cmb = fmaxf(cmb, __shfl_xor(cmb, 8));
            const float mnew = fmaxf(mrow[i], cmb);
            const float corr = __expf(mrow[i] - mnew);
            const float p0   = __expf(S[0][i] - mnew);
            const float p1   = __expf(S[1][i] - mnew);
            float rs = p0 + p1;
            rs += __shfl_xor(rs, 1);
            rs += __shfl_xor(rs, 2);
            rs += __shfl_xor(rs, 4);
            rs += __shfl_xor(rs, 8);
            lrow[i] = lrow[i] * corr + rs;
            mrow[i] = mnew;
            #pragma unroll
            for (int d = 0; d < 4; ++d) acc[d][i] *= corr;
            const int row = i + 8 * g;
            pw[row * 32 + lo]      = (_Float16)p0;
            pw[row * 32 + 16 + lo] = (_Float16)p1;
        }
        // per-wave LDS round trip: LDS is in-order per wave, but stop the
        // compiler from hoisting the cross-lane reads above the stores.
        asm volatile("s_wait_dscnt 0" ::: "memory");

        // ---- P (A layout) from LDS, then P @ V over 4 dh tiles ----
        V16H ap;
        const _Float16* pr = &pw[lo * 32];
        #pragma unroll
        for (int v = 0; v < 8; ++v) {
            const int K0 = 2 * v + 16 * g;
            ap.u[v] = *(const unsigned*)&pr[K0];
        }
        #pragma unroll
        for (int d = 0; d < 4; ++d) {
            V16H bv_;
            const _Float16* vr2 = &vl[(d * 16 + lo) * 32 + 16 * g];
            bv_.q[0] = ((const u32x4*)vr2)[0];
            bv_.q[1] = ((const u32x4*)vr2)[1];
            acc[d] = __builtin_amdgcn_wmma_f32_16x16x32_f16(
                false, ap.v, false, bv_.v, (short)0, acc[d], false, false);
        }
        __syncthreads();  // everyone done reading buf[cur] before it refills
    }

    // ---- normalize and scatter through inverse head-reshape ----
    const int b = bh >> 3, h = bh & 7;
    #pragma unroll
    for (int d = 0; d < 4; ++d) {
        const int dhg = d * 16 + lo;
        #pragma unroll
        for (int i = 0; i < 8; ++i) {
            const int nn = qbase + i + 8 * g;
            const float y = acc[d][i] / lrow[i];
            const int n  = h * 128 + (nn >> 3);
            const int dd = ((nn & 7) << 6) | dhg;
            out[((size_t)(b * 1024 + n)) * 512 + dd] = y;
        }
    }
}

extern "C" void kernel_launch(void* const* d_in, const int* in_sizes, int n_in,
                              void* d_out, int out_size, void* d_ws, size_t ws_size,
                              hipStream_t stream) {
    const float* x  = (const float*)d_in[0];
    const float* Wq = (const float*)d_in[1];
    const float* bq = (const float*)d_in[2];
    const float* Wk = (const float*)d_in[3];
    const float* bk = (const float*)d_in[4];
    const float* Wv = (const float*)d_in[5];
    const float* bv = (const float*)d_in[6];

    _Float16* qws = (_Float16*)d_ws;
    _Float16* kws = qws + (size_t)TOK_ * D_;
    _Float16* vws = kws + (size_t)TOK_ * D_;

    dim3 gp(TOK_ / 128, D_ / 64, 3);
    qkv_proj_kernel<<<gp, 256, 0, stream>>>(x, Wq, bq, Wk, bk, Wv, bv,
                                            qws, kws, vws);
    dim3 ga(B_ * H_, N_ / 128);
    attn_kernel<<<ga, 256, 0, stream>>>(qws, kws, vws, (float*)d_out);
}
